// KimiDeltaAttention_29154238005744
// MI455X (gfx1250) — matrix operand order repaired and checked by reading, not compiled, not run on hardware
//
#include <hip/hip_runtime.h>
#include <hip/hip_bf16.h>

// ---------------------------------------------------------------------------
// KDA (Kimi Delta Attention) forward for gfx1250 (MI455X), bf16 WMMA pipeline.
// ---------------------------------------------------------------------------

#define Hh    16
#define Dd    128
#define HD    2048      // H*D == HID
#define HIDc  2048
#define CHUNK 64

typedef __bf16 bf16_t;
typedef bf16_t bf16x16 __attribute__((ext_vector_type(16)));
typedef bf16_t bf16x4  __attribute__((ext_vector_type(4)));
typedef float  f32x8   __attribute__((ext_vector_type(8)));
typedef float  f32x4v  __attribute__((ext_vector_type(4)));
typedef unsigned int u32x4 __attribute__((ext_vector_type(4)));
typedef unsigned int u32x8 __attribute__((ext_vector_type(8)));

__device__ __forceinline__ f32x8 zero8() {
  f32x8 v;
#pragma unroll
  for (int i = 0; i < 8; ++i) v[i] = 0.f;
  return v;
}

__device__ __forceinline__ bf16x4 cvt_bf16x4(f32x4v v) {
  bf16x4 r;
#pragma unroll
  for (int i = 0; i < 4; ++i) r[i] = (bf16_t)v[i];
  return r;
}

__device__ __forceinline__ f32x8 wmma_bf16(bf16x16 a, bf16x16 b, f32x8 c) {
  return __builtin_amdgcn_wmma_f32_16x16x32_bf16(false, a, false, b, (short)0, c,
                                                 false, false);
}

// A fragment (16x32, 16-bit): lanes 0-15 rows M, K=0..15;
// lanes 16-31 same rows, K=16..31. src row-major MxK (contiguous K).
template <typename T>
__device__ __forceinline__ bf16x16 load_frag_A(const T* src, int ld, int m0,
                                               int k0, int lane) {
  bf16x16 f;
  int m  = m0 + (lane & 15);
  int kb = k0 + ((lane & 16) ? 16 : 0);
  const T* p = src + (long)m * ld + kb;
#pragma unroll
  for (int e = 0; e < 8; ++e) f[e] = (bf16_t)(float)p[e];
#pragma unroll
  for (int e = 0; e < 8; ++e) f[8 + e] = (bf16_t)(float)p[8 + e];
  return f;
}

// B fragment (32x16, 16-bit): lane = column N; lanes 0-15 hold K=0..15,
// lanes 16-31 hold K=16..31.  Source stored N-major (B^T), contiguous K.
template <typename T>
__device__ __forceinline__ bf16x16 load_frag_BT(const T* src, int ld, int k0,
                                                int n0, int lane) {
  bf16x16 f;
  int n  = n0 + (lane & 15);
  int kb = k0 + ((lane & 16) ? 16 : 0);
  const T* p = src + (long)n * ld + kb;
#pragma unroll
  for (int e = 0; e < 16; ++e) f[e] = (bf16_t)(float)p[e];
  return f;
}

// ---------------------------------------------------------------------------
// Tensor Data Mover: DMA a 2D fp32 tile (tile_d0 x tile_d1, row stride
// stride_el elements) from global memory into LDS.  D# built in SGPR tuples;
// group2/3 are zero-filled (both non-NULL => up-to-5D form, unused dims 0).
// Issue from a single wave; follow with s_wait_tensorcnt + workgroup barrier.
// ---------------------------------------------------------------------------
__device__ __forceinline__ void tdm_load_2d_f32(unsigned lds_off,
                                                const void* gptr,
                                                unsigned tile_d0, unsigned tile_d1,
                                                unsigned tensor_d0,
                                                unsigned tensor_d1,
                                                unsigned stride_el) {
  unsigned long long ga = (unsigned long long)gptr;
  u32x4 g0;
  g0[0] = 1u;                                              // count=1 (valid D#)
  g0[1] = lds_off;                                         // lds_addr (bytes)
  g0[2] = (unsigned)ga;                                    // global_addr[31:0]
  g0[3] = (unsigned)((ga >> 32) & 0x01FFFFFFu) | (2u << 30);  // [56:32], type=2
  u32x8 g1;
  g1[0] = (2u << 16);                                      // data_size=4B
  g1[1] = (tensor_d0 & 0xFFFFu) << 16;                     // tensor_dim0[15:0]
  g1[2] = (tensor_d0 >> 16) | ((tensor_d1 & 0xFFFFu) << 16);
  g1[3] = (tensor_d1 >> 16) | (tile_d0 << 16);             // tile_dim0
  g1[4] = tile_d1;                                         // tile_dim1 (dim2=0)
  g1[5] = stride_el;                                       // dim0_stride[31:0]
  g1[6] = 0u;
  g1[7] = 0u;
  u32x4 gz = {0u, 0u, 0u, 0u};
  asm volatile("tensor_load_to_lds %0, %1, %2, %3"
               :
               : "s"(g0), "s"(g1), "s"(gz), "s"(gz)
               : "memory");
}

// ---------------------------------------------------------------------------
// Tile staging for the GEMM: fp32 global -> bf16 LDS.
// Fast path: float4 (global_load_b128) + bf16x4 (ds_store_b64); B transposed
// via per-thread 4x4 micro-tiles. AL=true removes all bounds checks.
// ---------------------------------------------------------------------------
template <bool AL>
__device__ __forceinline__ void stage_tiles(const float* __restrict__ A,
                                            const float* __restrict__ Bw,
                                            int M, int N, int K, int bm, int bn,
                                            int kk, int tid,
                                            bf16_t (*As)[40], bf16_t (*Bs)[40]) {
  const bool kfull = AL || (kk + 32 <= K);
  if (AL || ((bm + 128 <= M) && kfull)) {
#pragma unroll
    for (int i4 = tid; i4 < 1024; i4 += 256) {
      int r = i4 >> 3, c0 = (i4 & 7) * 4;
      f32x4v vv = *(const f32x4v*)(A + (long)(bm + r) * K + kk + c0);
      *(bf16x4*)&As[r][c0] = cvt_bf16x4(vv);
    }
  } else {
    for (int i = tid; i < 128 * 32; i += 256) {
      int r = i >> 5, c = i & 31;
      int gr = bm + r, gc = kk + c;
      As[r][c] = (bf16_t)((gr < M && gc < K) ? A[(long)gr * K + gc] : 0.f);
    }
  }
  if (AL || ((bn + 128 <= N) && kfull)) {
    int r0 = (tid >> 5) * 4, c0 = (tid & 31) * 4;
    f32x4v rows[4];
#pragma unroll
    for (int j = 0; j < 4; ++j)
      rows[j] = *(const f32x4v*)(Bw + (long)(kk + r0 + j) * N + bn + c0);
#pragma unroll
    for (int cc = 0; cc < 4; ++cc) {
      bf16x4 col;
#pragma unroll
      for (int j = 0; j < 4; ++j) col[j] = (bf16_t)rows[j][cc];
      *(bf16x4*)&Bs[c0 + cc][r0] = col;
    }
  } else {
    for (int i = tid; i < 32 * 128; i += 256) {
      int r = i >> 7, c = i & 127;
      int gr = kk + r, gc = bn + c;
      Bs[c][r] = (bf16_t)((gr < K && gc < N) ? Bw[(long)gr * N + gc] : 0.f);
    }
  }
}

// ---------------------------------------------------------------------------
// Generic GEMM: C[M,N] = A[M,K] @ B[K,N], fp32 in/out, bf16 WMMA inside.
// Block tile 128x128, K-step 32, double-buffered LDS, 8 waves (wave32).
// AL=true: all dims assumed multiples of the tile (no checks anywhere).
// ---------------------------------------------------------------------------
template <bool AL>
__global__ __launch_bounds__(256) void kda_gemm_kernel(
    const float* __restrict__ A, const float* __restrict__ Bw,
    float* __restrict__ C, int M, int N, int K) {
  __shared__ bf16_t As[2][128][40];   // [m][k]
  __shared__ bf16_t Bs[2][128][40];   // [n][k] (transposed)
  const int bm = blockIdx.y * 128, bn = blockIdx.x * 128;
  const int tid = threadIdx.x, lane = tid & 31, wave = tid >> 5;
  const int wr = wave >> 1, wc = wave & 1;

  f32x8 acc[2][4];
#pragma unroll
  for (int i = 0; i < 2; ++i)
#pragma unroll
    for (int j = 0; j < 4; ++j) acc[i][j] = zero8();

  const int nk = (K + 31) / 32;
  stage_tiles<AL>(A, Bw, M, N, K, bm, bn, 0, tid, As[0], Bs[0]);
  __syncthreads();

  for (int t = 0; t < nk; ++t) {
    const int cur = t & 1;
    if (t + 1 < nk)
      stage_tiles<AL>(A, Bw, M, N, K, bm, bn, (t + 1) * 32, tid, As[cur ^ 1],
                      Bs[cur ^ 1]);

    bf16x16 af0 = load_frag_A(&As[cur][0][0], 40, wr * 32, 0, lane);
    bf16x16 af1 = load_frag_A(&As[cur][0][0], 40, wr * 32 + 16, 0, lane);
#pragma unroll
    for (int nt = 0; nt < 4; ++nt) {
      bf16x16 bf = load_frag_BT(&Bs[cur][0][0], 40, 0, wc * 64 + nt * 16, lane);
      acc[0][nt] = wmma_bf16(af0, bf, acc[0][nt]);
      acc[1][nt] = wmma_bf16(af1, bf, acc[1][nt]);
    }
    __syncthreads();
  }

#pragma unroll
  for (int mt = 0; mt < 2; ++mt)
#pragma unroll
    for (int nt = 0; nt < 4; ++nt) {
      int col = bn + wc * 64 + nt * 16 + (lane & 15);
      int rb  = bm + wr * 32 + mt * 16 + ((lane & 16) ? 8 : 0);
      if (AL || col < N) {
#pragma unroll
        for (int e = 0; e < 8; ++e)
          if (AL || rb + e < M) C[(long)(rb + e) * N + col] = acc[mt][nt][e];
      }
    }
}

// ---------------------------------------------------------------------------
// Causal depthwise conv (K=4) + SiLU (+ optional per-head L2 norm over D).
// ---------------------------------------------------------------------------
__global__ __launch_bounds__(128) void kda_conv_kernel(
    const float* __restrict__ pre, const float* __restrict__ kern,
    float* __restrict__ out, int Sd, int do_norm) {
  int bs = blockIdx.x;
  int b = bs / Sd, s = bs % Sd;
  int h = blockIdx.y, d = threadIdx.x;
  int c = h * Dd + d;
  long base = ((long)b * Sd + s) * HD + c;

  float acc = 0.f;
#pragma unroll
  for (int t = 0; t < 4; ++t) {
    int sp = s - 3 + t;
    if (sp >= 0) acc += pre[((long)b * Sd + sp) * HD + c] * kern[t * HD + c];
  }
  float y = acc / (1.f + __expf(-acc));  // silu

  if (do_norm) {
    __shared__ float red[128];
    red[d] = y * y;
    __syncthreads();
    for (int off = 64; off > 0; off >>= 1) {
      if (d < off) red[d] += red[d + off];
      __syncthreads();
    }
    y *= rsqrtf(red[0] + 1e-6f);
  }
  out[base] = y;
}

// ---------------------------------------------------------------------------
// Small elementwise kernels.
// ---------------------------------------------------------------------------
__global__ void kda_sigmoid_kernel(float* __restrict__ p, long n) {
  long i = (long)blockIdx.x * blockDim.x + threadIdx.x;
  if (i < n) p[i] = 1.f / (1.f + __expf(-p[i]));
}

__global__ void kda_gates_kernel(float* __restrict__ g_lin,
                                 const float* __restrict__ A_log,
                                 const float* __restrict__ dt_bias, long n) {
  long i = (long)blockIdx.x * blockDim.x + threadIdx.x;
  if (i >= n) return;
  int c = (int)(i & (HD - 1));
  int h = c >> 7;
  float x  = g_lin[i] + dt_bias[c];
  float sp = (x > 20.f) ? x : log1pf(__expf(x));  // softplus
  g_lin[i] = -__expf(A_log[h]) * sp;
}

__global__ __launch_bounds__(128) void kda_combine_kernel(
    const float* __restrict__ core, const float* __restrict__ gate_lin,
    const float* __restrict__ bgb, const float* __restrict__ rms_scale,
    float* __restrict__ y) {
  long bsh = blockIdx.x;
  int d = threadIdx.x;
  long idx = bsh * Dd + d;
  float cf = core[idx];
  __shared__ float red[128];
  red[d] = cf * cf;
  __syncthreads();
  for (int off = 64; off > 0; off >>= 1) {
    if (d < off) red[d] += red[d + off];
    __syncthreads();
  }
  float normed = cf * rsqrtf(red[0] * (1.f / 128.f) + 1e-5f) * rms_scale[d];
  int c = (int)(idx & (HD - 1));
  float gv = gate_lin[idx] + bgb[c];
  y[idx] = normed * (1.f / (1.f + __expf(-gv)));
}

// ---------------------------------------------------------------------------
// Chunked gated delta-rule core. One workgroup (256 threads / 8 waves) per
// (b,h); ~246KB dynamic LDS (CDNA5: up to 320KB per workgroup).
// g-tile is DMA'd into LDS by the Tensor Data Mover (TENSORcnt-tracked).
// ---------------------------------------------------------------------------
__global__ __launch_bounds__(256) void kda_chunk_kernel(
    const float* __restrict__ q, const float* __restrict__ k,
    const float* __restrict__ v, const float* __restrict__ g,
    const float* __restrict__ beta, float* __restrict__ core, int Sd) {
  extern __shared__ char smem_raw[];
  float*  ST    = (float*)smem_raw;            // 128x128 state transposed
  float*  gt    = ST + 128 * 128;              // 64x128 cumulative gate
  float*  Amat  = gt + CHUNK * 128;            // 64x64
  float*  Tmat  = Amat + CHUNK * CHUNK;        // 64x64
  float*  betas = Tmat + CHUNK * CHUNK;        // 64
  bf16_t* qg    = (bf16_t*)(betas + CHUNK);    // 64x128 q*scale*e^g   (A)
  bf16_t* kg    = qg + CHUNK * 128;            // 64x128 k*e^g         (A)
  bf16_t* kgT   = kg + CHUNK * 128;            // 128x64 k*e^g         (B)
  bf16_t* knt   = kgT + 128 * CHUNK;           // 64x128 kng -> 128x64 k_tail^T
  bf16_t* vT    = knt + CHUNK * 128;           // 128x64 v / v_new     (B)
  bf16_t* wv    = vT + 128 * CHUNK;            // 64x128 -(Tf@kg)      (A)
  bf16_t* Tb    = wv + CHUNK * 128;            // 64x64  Tf            (A)
  bf16_t* attnb = Tb + CHUNK * CHUNK;          // 64x64  masked attn   (A)

  const int b = blockIdx.x / Hh, h = blockIdx.x % Hh;
  const int tid = threadIdx.x, lane = tid & 31, wave = tid >> 5;
  const float scale = 0.08838834764831845f;  // 128^-0.5

  for (int i = tid; i < 128 * 128; i += 256) ST[i] = 0.f;
  __syncthreads();

  const int nchunks = Sd / CHUNK;
  for (int ci = 0; ci < nchunks; ++ci) {
    const long base = ((long)b * Sd + (long)ci * CHUNK) * HD + (long)h * Dd;

    // 1) TDM: async DMA of the 64x128 fp32 g-tile into LDS (wave 0 issues),
    //    beta loaded normally; barrier releases consumers.
    if (wave == 0) {
      tdm_load_2d_f32((unsigned)(unsigned long long)gt, (const void*)(g + base),
                      /*tile_d0=*/128u, /*tile_d1=*/64u,
                      /*tensor_d0=*/128u, /*tensor_d1=*/64u,
                      /*stride_el=*/(unsigned)HD);
      __builtin_amdgcn_s_wait_tensorcnt(0);
    }
    if (tid < CHUNK) betas[tid] = beta[((long)b * Sd + ci * CHUNK + tid) * Hh + h];
    __syncthreads();

    // 2) in-chunk cumulative sum along rows (per column d)
    if (tid < 128) {
      float a = 0.f;
      for (int r = 0; r < CHUNK; ++r) {
        a += gt[r * 128 + tid];
        gt[r * 128 + tid] = a;
      }
    }
    __syncthreads();

    // 3) derived bf16 operand tiles (vectorized loads, b64 row-major stores)
    for (int i4 = tid; i4 < CHUNK * 32; i4 += 256) {
      int r = i4 >> 5, c0 = (i4 & 31) * 4;
      long gi = base + (long)r * HD + c0;
      f32x4v qv = *(const f32x4v*)(q + gi);
      f32x4v kv = *(const f32x4v*)(k + gi);
      f32x4v vv = *(const f32x4v*)(v + gi);
      bf16x4 qq, kk4, kn4;
#pragma unroll
      for (int j = 0; j < 4; ++j) {
        float gc = gt[r * 128 + c0 + j];
        float eg = __expf(gc);
        qq[j] = (bf16_t)(qv[j] * scale * eg);
        bf16_t kge = (bf16_t)(kv[j] * eg);
        kk4[j] = kge;
        kn4[j] = (bf16_t)(kv[j] * __expf(-gc));
        kgT[(c0 + j) * CHUNK + r] = kge;
        vT[(c0 + j) * CHUNK + r] = (bf16_t)vv[j];
      }
      *(bf16x4*)&qg[r * 128 + c0]  = qq;
      *(bf16x4*)&kg[r * 128 + c0]  = kk4;
      *(bf16x4*)&knt[r * 128 + c0] = kn4;
    }
    __syncthreads();

    // 4) A = strict_mask(kg @ kng^T) * beta_row ; attn = incl_mask(qg @ kng^T)
#pragma unroll
    for (int p = 0; p < 2; ++p) {
      int t = wave * 2 + p, mt = t >> 2, nt = t & 3;
      f32x8 aA = zero8(), aT = zero8();
#pragma unroll
      for (int kt2 = 0; kt2 < 4; ++kt2) {
        bf16x16 bfr = load_frag_BT(knt, 128, kt2 * 32, nt * 16, lane);
        aA = wmma_bf16(load_frag_A(kg, 128, mt * 16, kt2 * 32, lane), bfr, aA);
        aT = wmma_bf16(load_frag_A(qg, 128, mt * 16, kt2 * 32, lane), bfr, aT);
      }
      int col = nt * 16 + (lane & 15);
      int rb  = mt * 16 + ((lane & 16) ? 8 : 0);
#pragma unroll
      for (int e = 0; e < 8; ++e) {
        int row = rb + e;
        Amat[row * CHUNK + col]  = (row > col) ? aA[e] * betas[row] : 0.f;
        attnb[row * CHUNK + col] = (bf16_t)((row >= col) ? aT[e] : 0.f);
      }
    }
    __syncthreads();

    // 5a) init T = I ; overwrite knt with k_tail^T [d][i] = k*e^{g_last - g}
    for (int i = tid; i < CHUNK * CHUNK; i += 256) {
      int r = i >> 6, c = i & 63;
      Tmat[i] = (r == c) ? 1.f : 0.f;
    }
    for (int i4 = tid; i4 < CHUNK * 32; i4 += 256) {
      int r = i4 >> 5, c0 = (i4 & 31) * 4;
      f32x4v kv = *(const f32x4v*)(k + base + (long)r * HD + c0);
#pragma unroll
      for (int j = 0; j < 4; ++j) {
        int c = c0 + j;
        knt[c * CHUNK + r] =
            (bf16_t)(kv[j] * __expf(gt[(CHUNK - 1) * 128 + c] - gt[r * 128 + c]));
      }
    }
    __syncthreads();

    // 5b) T = (I + A)^{-1} by forward substitution, one thread per column
    if (tid < CHUNK) {
      int c = tid;
      for (int r = c + 1; r < CHUNK; ++r) {
        float s = 0.f;
        for (int j = c; j < r; ++j) s += Amat[r * CHUNK + j] * Tmat[j * CHUNK + c];
        Tmat[r * CHUNK + c] = -s;
      }
    }
    __syncthreads();

    // 5c) Tf = T * beta_col -> bf16
    for (int i = tid; i < CHUNK * CHUNK; i += 256)
      Tb[i] = (bf16_t)(Tmat[i] * betas[i & 63]);
    __syncthreads();

    // 6) U = Tf@v (regs) ; wv = -(Tf@kg) -> LDS
    f32x8 Uf[4];
    int tm[4], tn[4];
#pragma unroll
    for (int p = 0; p < 4; ++p) {
      int t = wave * 4 + p;
      tm[p] = t >> 3;
      tn[p] = t & 7;
      f32x8 aU = zero8(), aW = zero8();
#pragma unroll
      for (int kt2 = 0; kt2 < 2; ++kt2) {
        bf16x16 af = load_frag_A(Tb, CHUNK, tm[p] * 16, kt2 * 32, lane);
        aU = wmma_bf16(af, load_frag_BT(vT, CHUNK, kt2 * 32, tn[p] * 16, lane), aU);
        aW = wmma_bf16(af, load_frag_BT(kgT, CHUNK, kt2 * 32, tn[p] * 16, lane), aW);
      }
      Uf[p] = aU;
      int col = tn[p] * 16 + (lane & 15);
      int rb  = tm[p] * 16 + ((lane & 16) ? 8 : 0);
#pragma unroll
      for (int e = 0; e < 8; ++e) wv[(rb + e) * 128 + col] = (bf16_t)(-aW[e]);
    }
    __syncthreads();

    // 7) v_new = U + wv@S  (B = S via ST, fp32 -> bf16 per fragment)
#pragma unroll
    for (int p = 0; p < 4; ++p) {
      f32x8 acc = Uf[p];
#pragma unroll
      for (int kt2 = 0; kt2 < 4; ++kt2)
        acc = wmma_bf16(load_frag_A(wv, 128, tm[p] * 16, kt2 * 32, lane),
                        load_frag_BT(ST, 128, kt2 * 32, tn[p] * 16, lane), acc);
      int col = tn[p] * 16 + (lane & 15);
      int rb  = tm[p] * 16 + ((lane & 16) ? 8 : 0);
#pragma unroll
      for (int e = 0; e < 8; ++e) vT[col * CHUNK + (rb + e)] = (bf16_t)acc[e];
    }
    __syncthreads();

    // 8) O = attn@v_new + qg@S -> global
#pragma unroll
    for (int p = 0; p < 4; ++p) {
      f32x8 acc = zero8();
#pragma unroll
      for (int kt2 = 0; kt2 < 2; ++kt2)
        acc = wmma_bf16(load_frag_A(attnb, CHUNK, tm[p] * 16, kt2 * 32, lane),
                        load_frag_BT(vT, CHUNK, kt2 * 32, tn[p] * 16, lane), acc);
#pragma unroll
      for (int kt2 = 0; kt2 < 4; ++kt2)
        acc = wmma_bf16(load_frag_A(qg, 128, tm[p] * 16, kt2 * 32, lane),
                        load_frag_BT(ST, 128, kt2 * 32, tn[p] * 16, lane), acc);
      int col = tn[p] * 16 + (lane & 15);
      int rb  = tm[p] * 16 + ((lane & 16) ? 8 : 0);
#pragma unroll
      for (int e = 0; e < 8; ++e)
        core[base + (long)(rb + e) * HD + col] = acc[e];
    }
    __syncthreads();

    // 9) S = S * e^{g_last,row} + k_tail^T @ v_new   (updates ST in place)
#pragma unroll
    for (int p = 0; p < 8; ++p) {
      int t = wave * 8 + p, mt2 = t >> 3, nt2 = t & 7;
      f32x8 acc = zero8();
#pragma unroll
      for (int kt2 = 0; kt2 < 2; ++kt2)
        acc = wmma_bf16(load_frag_A(knt, CHUNK, mt2 * 16, kt2 * 32, lane),
                        load_frag_BT(vT, CHUNK, kt2 * 32, nt2 * 16, lane), acc);
      int col = nt2 * 16 + (lane & 15);            // e (dv index)
      int rb  = mt2 * 16 + ((lane & 16) ? 8 : 0);  // d (dk index)
#pragma unroll
      for (int e = 0; e < 8; ++e) {
        int row = rb + e;
        ST[col * 128 + row] =
            ST[col * 128 + row] * __expf(gt[(CHUNK - 1) * 128 + row]) + acc[e];
      }
    }
    __syncthreads();
  }
}

// ---------------------------------------------------------------------------
// Host-side launch sequence.
// ---------------------------------------------------------------------------
extern "C" void kernel_launch(void* const* d_in, const int* in_sizes, int n_in,
                              void* d_out, int out_size, void* d_ws,
                              size_t ws_size, hipStream_t stream) {
  (void)in_sizes; (void)n_in; (void)out_size; (void)ws_size;
  const float* x        = (const float*)d_in[0];
  const float* Wq       = (const float*)d_in[1];
  const float* Wk       = (const float*)d_in[2];
  const float* Wv       = (const float*)d_in[3];
  const float* cq       = (const float*)d_in[4];
  const float* ck       = (const float*)d_in[5];
  const float* cv       = (const float*)d_in[6];
  const float* Wb       = (const float*)d_in[7];
  const float* Wfa      = (const float*)d_in[8];
  const float* Wfb      = (const float*)d_in[9];
  const float* Wga      = (const float*)d_in[10];
  const float* Wgb      = (const float*)d_in[11];
  const float* bgb      = (const float*)d_in[12];
  const float* A_log    = (const float*)d_in[13];
  const float* dt_bias  = (const float*)d_in[14];
  const float* rms      = (const float*)d_in[15];
  const float* Wo       = (const float*)d_in[16];
  float* out = (float*)d_out;

  constexpr int Bc = 2, Sc = 4096;
  const int M = Bc * Sc;  // 8192

  const size_t sz_big = (size_t)M * HD * sizeof(float);
  char* w = (char*)d_ws;
  float* buf0  = (float*)(w + 0 * sz_big);  // pre_q -> g_lin/g -> y
  float* buf1  = (float*)(w + 1 * sz_big);  // pre_k -> gate_lin
  float* buf2  = (float*)(w + 2 * sz_big);  // pre_v -> core
  float* qb    = (float*)(w + 3 * sz_big);
  float* kb    = (float*)(w + 4 * sz_big);
  float* vb    = (float*)(w + 5 * sz_big);
  float* lowr  = (float*)(w + 6 * sz_big);                       // M x 128
  float* betab = (float*)(w + 6 * sz_big + (size_t)M * Dd * sizeof(float));

  dim3 blk256(256);
  auto ggrid = [](int M_, int N_) {
    return dim3((unsigned)((N_ + 127) / 128), (unsigned)((M_ + 127) / 128));
  };

  // 1-3) projections (aligned fast GEMM)
  kda_gemm_kernel<true><<<ggrid(M, HD), blk256, 0, stream>>>(x, Wq, buf0, M, HD, HIDc);
  kda_gemm_kernel<true><<<ggrid(M, HD), blk256, 0, stream>>>(x, Wk, buf1, M, HD, HIDc);
  kda_gemm_kernel<true><<<ggrid(M, HD), blk256, 0, stream>>>(x, Wv, buf2, M, HD, HIDc);

  // 4-6) causal depthwise conv + silu (+ l2norm for q,k)
  kda_conv_kernel<<<dim3((unsigned)M, Hh), dim3(128), 0, stream>>>(buf0, cq, qb, Sc, 1);
  kda_conv_kernel<<<dim3((unsigned)M, Hh), dim3(128), 0, stream>>>(buf1, ck, kb, Sc, 1);
  kda_conv_kernel<<<dim3((unsigned)M, Hh), dim3(128), 0, stream>>>(buf2, cv, vb, Sc, 0);

  // 7-8) beta = sigmoid(x @ Wb)  (generic GEMM: N=16)
  kda_gemm_kernel<false><<<ggrid(M, Hh), blk256, 0, stream>>>(x, Wb, betab, M, Hh, HIDc);
  {
    long n = (long)M * Hh;
    kda_sigmoid_kernel<<<(unsigned)((n + 255) / 256), blk256, 0, stream>>>(betab, n);
  }

  // 9-11) g = -exp(A_log) * softplus(x @ Wfa @ Wfb + dt_bias)  (into buf0)
  kda_gemm_kernel<true><<<ggrid(M, Dd), blk256, 0, stream>>>(x, Wfa, lowr, M, Dd, HIDc);
  kda_gemm_kernel<true><<<ggrid(M, HD), blk256, 0, stream>>>(lowr, Wfb, buf0, M, HD, Dd);
  {
    long n = (long)M * HD;
    kda_gates_kernel<<<(unsigned)((n + 255) / 256), blk256, 0, stream>>>(buf0, A_log,
                                                                         dt_bias, n);
  }

  // 12-13) gate_lin = x @ Wga @ Wgb  (into buf1; +bgb folded into combine)
  kda_gemm_kernel<true><<<ggrid(M, Dd), blk256, 0, stream>>>(x, Wga, lowr, M, Dd, HIDc);
  kda_gemm_kernel<true><<<ggrid(M, HD), blk256, 0, stream>>>(lowr, Wgb, buf1, M, HD, Dd);

  // 14) chunked delta-rule core -> buf2
  {
    size_t lds = (size_t)(128 * 128 + CHUNK * 128 + 2 * CHUNK * CHUNK + CHUNK) *
                     sizeof(float) +
                 (size_t)(6 * CHUNK * 128 + 2 * CHUNK * CHUNK) * sizeof(bf16_t);
    kda_chunk_kernel<<<dim3(Bc * Hh), blk256, lds, stream>>>(qb, kb, vb, buf0, betab,
                                                             buf2, Sc);
  }

  // 15) rmsnorm * sigmoid(gate) -> buf0
  kda_combine_kernel<<<dim3((unsigned)(M * Hh)), dim3(128), 0, stream>>>(
      buf2, buf1, bgb, rms, buf0);

  // 16) output projection
  kda_gemm_kernel<true><<<ggrid(M, HIDc), blk256, 0, stream>>>(buf0, Wo, out, M, HIDc, HD);
}
